// EGNN_9371618639972
// MI455X (gfx1250) — matrix operand order repaired and checked by reading, not compiled, not run on hardware
//
#include <hip/hip_runtime.h>

typedef __attribute__((ext_vector_type(16))) __bf16 v16bf;
typedef __attribute__((ext_vector_type(8)))  float  v8f;
typedef __attribute__((ext_vector_type(4)))  int    v4i;

union Frag { v16bf bf; v4i i2[2]; };

__device__ __forceinline__ float silu_f(float x) {
    return x / (1.0f + __expf(-x));
}

// ---------------------------------------------------------------------------
// radial[e] = || x[row[e]] - x[col[e]] ||^2
// ---------------------------------------------------------------------------
__global__ void radial_kernel(const float* __restrict__ x,
                              const int* __restrict__ row, const int* __restrict__ col,
                              float* __restrict__ radial, int E) {
    int e = blockIdx.x * blockDim.x + threadIdx.x;
    if (e >= E) return;
    int r = row[e], c = col[e];
    float dx = x[r * 3 + 0] - x[c * 3 + 0];
    float dy = x[r * 3 + 1] - x[c * 3 + 1];
    float dz = x[r * 3 + 2] - x[c * 3 + 2];
    radial[e] = dx * dx + dy * dy + dz * dz;
}

// ---------------------------------------------------------------------------
// h = h0 @ emb_w + emb_b   (N x 11 @ 11 x 128) ; also mirror h in bf16
// ---------------------------------------------------------------------------
__global__ void embed_kernel(const float* __restrict__ h0,
                             const float* __restrict__ w, const float* __restrict__ b,
                             float* __restrict__ h, __bf16* __restrict__ hbf, int Nn) {
    int idx = blockIdx.x * blockDim.x + threadIdx.x;
    if (idx >= Nn * 128) return;
    int n = idx >> 7, f = idx & 127;
    float acc = b[f];
#pragma unroll
    for (int k = 0; k < 11; ++k) acc += h0[n * 11 + k] * w[k * 128 + f];
    h[idx] = acc;
    hbf[idx] = (__bf16)acc;
}

// ---------------------------------------------------------------------------
// Re-pack a (layers, Ksrc, 128) f32 weight into wave32 bf16 WMMA B-fragments,
// K zero-padded to 32*Ksteps. Per (layer,kstep,nblock): 32 lanes x 16 bf16
// contiguous -> B loads are two aligned b128 per lane.
//   lane L holds N = nb*16 + (L&15),  K = 32*ks + (L>>4)*8 + (j&7) + (j>>3)*16  (j = 0..15)
// ---------------------------------------------------------------------------
__global__ void fragw_kernel(const float* __restrict__ src, __bf16* __restrict__ dst,
                             int Ksrc, int Ksteps, int layers) {
    int idx = blockIdx.x * blockDim.x + threadIdx.x;
    int total = layers * Ksteps * 8 * 512;
    if (idx >= total) return;
    int j    = idx & 15;
    int lane = (idx >> 4) & 31;
    int nb   = (idx >> 9) & 7;
    int rest = idx >> 12;
    int ks   = rest % Ksteps;
    int l    = rest / Ksteps;
    int K = 32 * ks + (lane >> 4) * 8 + (j & 7) + (j >> 3) * 16;
    int Ncol = nb * 16 + (lane & 15);
    float v = (K < Ksrc) ? src[(l * Ksrc + K) * 128 + Ncol] : 0.0f;
    dst[idx] = (__bf16)v;
}

__global__ void zero_kernel(float* __restrict__ p, int n) {
    int i = blockIdx.x * blockDim.x + threadIdx.x;
    if (i < n) p[i] = 0.0f;
}

// ---------------------------------------------------------------------------
// A-fragment (16x32 bf16) from row-major LDS tile (row stride ldw bf16).
// lane<16 : row=lane,    K = {k0..k0+7, k0+16..k0+23}
// lane>=16: row=lane-16, K = {k0+8..k0+15, k0+24..k0+31}
// Two aligned ds_load_b128 per fragment.
// ---------------------------------------------------------------------------
__device__ __forceinline__ Frag load_a_frag(const __bf16* base, int ldw,
                                            int k0, int lane) {
    int r16 = lane & 15, hh = lane >> 4;
    const __bf16* ap = base + r16 * ldw + k0 + hh * 8;
    Frag a;
    a.i2[0] = *(const v4i*)ap;
    a.i2[1] = *(const v4i*)(ap + 16);
    return a;
}

__device__ __forceinline__ void load_b_frag(Frag& b, const __bf16* wf, int ks,
                                            int wave, int lane) {
    const __bf16* bp = wf + (((ks * 8 + wave) * 32 + lane) << 4);
    b.i2[0] = ((const v4i*)bp)[0];
    b.i2[1] = ((const v4i*)bp)[1];
}

// Ping-pong K-loop + sched_group_barrier pipeline. Source-level double
// buffering alone is undone by the pre-RA scheduler (it sinks each ds_load
// pair to its consumer and RA merges the buffers -> s_wait_dscnt 0x0 per
// WMMA). The sched_group_barrier pipeline dictates issue order
//   DS_READ x4, (WMMA x1, DS_READ x2) x (K-2), WMMA x2
// which overlaps the two A buffers' lifetimes before RA, enabling partial
// dscnt waits (one WMMA of LDS latency hidden per step).
//   mask 0x100 = DS read, 0x008 = MFMA/WMMA.
template <int KSTEPS>
__device__ __forceinline__ v8f gemm_kloop(const __bf16* tile, int ldw, int lane,
                                          const Frag* B, v8f acc) {
    Frag buf[2];
    buf[0] = load_a_frag(tile, ldw, 0, lane);
    if (KSTEPS > 1) buf[1] = load_a_frag(tile, ldw, 32, lane);
#pragma unroll
    for (int ks = 0; ks < KSTEPS; ++ks) {
        acc = __builtin_amdgcn_wmma_f32_16x16x32_bf16(false, buf[ks & 1].bf, false,
                                                      B[ks].bf, (short)0, acc,
                                                      false, false);
        if (ks + 2 < KSTEPS)
            buf[ks & 1] = load_a_frag(tile, ldw, (ks + 2) * 32, lane);
    }
#if __has_builtin(__builtin_amdgcn_sched_group_barrier)
    __builtin_amdgcn_sched_group_barrier(0x100, 4, 0);          // 2 frags in flight
#pragma unroll
    for (int ks = 0; ks < KSTEPS - 2; ++ks) {
        __builtin_amdgcn_sched_group_barrier(0x008, 1, 0);      // WMMA ks
        __builtin_amdgcn_sched_group_barrier(0x100, 2, 0);      // prefetch ks+2
    }
    __builtin_amdgcn_sched_group_barrier(0x008, 2, 0);          // drain last two
#endif
    return acc;
}

// ---------------------------------------------------------------------------
// Edge MLP + scatter. One WG (256 thr = 8 waves) processes EDGE_TPW 16-edge
// tiles; all B-fragments (13 x 8 VGPRs) live in registers across the loop.
// Wave w computes output features [16w, 16w+16).
// ---------------------------------------------------------------------------
#define EDGE_TPW 16
#define NODE_TPW 5

__global__ __launch_bounds__(256)
void edge_kernel(const __bf16* __restrict__ hbf, const float* __restrict__ radial,
                 const float* __restrict__ edge_attr, const float* __restrict__ edge_mask,
                 const int* __restrict__ row, const int* __restrict__ col,
                 const __bf16* __restrict__ w1f, const float* __restrict__ b1,
                 const __bf16* __restrict__ w2f, const float* __restrict__ b2,
                 float* __restrict__ agg) {
    __shared__ __bf16 s_in[16 * 288];   // [h_row | h_col | radial | attr | pad]
    __shared__ __bf16 s_m[16 * 128];    // intermediate message tile (bf16)
    __shared__ int    s_row[16];

    const int t = threadIdx.x;
    const int lane = t & 31, wave = t >> 5;
    const int hh = lane >> 4, r16 = lane & 15;
    const int Ncol = wave * 16 + r16;
    const int ge = t >> 4, gc = t & 15;     // gather role: (edge, 32B chunk)

    // persistent weight fragments + biases for this wave's N-slice
    Frag B1[9], B2[4];
#pragma unroll
    for (int ks = 0; ks < 9; ++ks) load_b_frag(B1[ks], w1f, ks, wave, lane);
#pragma unroll
    for (int ks = 0; ks < 4; ++ks) load_b_frag(B2[ks], w2f, ks, wave, lane);
    const float bias1 = b1[Ncol];
    const float bias2 = b2[Ncol];

    for (int tt = 0; tt < EDGE_TPW; ++tt) {
        const int e0 = (blockIdx.x * EDGE_TPW + tt) * 16;
        __syncthreads();   // previous iteration's LDS reads complete
        {
            int ee = e0 + ge;
            int r = row[ee], cl = col[ee];
            const __bf16* src = (gc < 8) ? (hbf + (size_t)r * 128 + gc * 16)
                                         : (hbf + (size_t)cl * 128 + (gc - 8) * 16);
            __bf16* dst = &s_in[ge * 288 + ((gc < 8) ? gc * 16 : 128 + (gc - 8) * 16)];
            ((v4i*)dst)[0] = ((const v4i*)src)[0];
            ((v4i*)dst)[1] = ((const v4i*)src)[1];
            if (gc == 0) s_row[ge] = r;
        }
        if (t < 16) {  // tail: radial(1) + attr(4) + zero pad (27)
            int ee = e0 + t;
            __bf16* dst = &s_in[t * 288 + 256];
            dst[0] = (__bf16)radial[ee];
#pragma unroll
            for (int k = 0; k < 4; ++k) dst[1 + k] = (__bf16)edge_attr[ee * 4 + k];
#pragma unroll
            for (int k = 5; k < 32; ++k) dst[k] = (__bf16)0.0f;
        }
        __syncthreads();

        // GEMM1: 16x288 @ 288x128  (9 WMMA / wave, B in registers)
        v8f acc = {};
        acc = gemm_kloop<9>(s_in, 288, lane, B1, acc);
#pragma unroll
        for (int v = 0; v < 8; ++v) {
            int M = v + 8 * hh;
            s_m[M * 128 + Ncol] = (__bf16)silu_f(acc[v] + bias1);
        }
        __syncthreads();

        // GEMM2: 16x128 @ 128x128  (4 WMMA / wave)
        v8f acc2 = {};
        acc2 = gemm_kloop<4>(s_m, 128, lane, B2, acc2);
#pragma unroll
        for (int v = 0; v < 8; ++v) {
            int M = v + 8 * hh;
            float mv = silu_f(acc2[v] + bias2) * edge_mask[e0 + M];
            atomicAdd(&agg[(size_t)s_row[M] * 128 + Ncol], mv);
        }
    }
}

// ---------------------------------------------------------------------------
// Node MLP + residual. One WG per NODE_TPW 16-node tiles, weights in regs.
// node_in = [h(128) | agg(128) | h0(11) | pad] -> silu(.@nw1+nb1)@nw2+nb2; h += out
// ---------------------------------------------------------------------------
__global__ __launch_bounds__(256)
void node_kernel(float* __restrict__ h, const float* __restrict__ agg,
                 const float* __restrict__ h0,
                 const __bf16* __restrict__ w1f, const float* __restrict__ b1,
                 const __bf16* __restrict__ w2f, const float* __restrict__ b2,
                 __bf16* __restrict__ hbf) {
    __shared__ __bf16 s_in[16 * 288];
    __shared__ __bf16 s_m[16 * 128];

    const int t = threadIdx.x;
    const int lane = t & 31, wave = t >> 5;
    const int hh = lane >> 4, r16 = lane & 15;
    const int Ncol = wave * 16 + r16;
    const int gn = t >> 4, gc = t & 15;

    Frag B1[9], B2[4];
#pragma unroll
    for (int ks = 0; ks < 9; ++ks) load_b_frag(B1[ks], w1f, ks, wave, lane);
#pragma unroll
    for (int ks = 0; ks < 4; ++ks) load_b_frag(B2[ks], w2f, ks, wave, lane);
    const float bias1 = b1[Ncol];
    const float bias2 = b2[Ncol];

    for (int tt = 0; tt < NODE_TPW; ++tt) {
        const int n0 = (blockIdx.x * NODE_TPW + tt) * 16;
        __syncthreads();
        {
            size_t node = (size_t)(n0 + gn);
            const float* src = (gc < 8) ? (h + node * 128 + gc * 16)
                                        : (agg + node * 128 + (gc - 8) * 16);
            __bf16* dst = &s_in[gn * 288 + ((gc < 8) ? gc * 16 : 128 + (gc - 8) * 16)];
#pragma unroll
            for (int k = 0; k < 16; ++k) dst[k] = (__bf16)src[k];
        }
        if (t < 16) {  // tail: h0 (11) + zero pad (21)
            size_t node = (size_t)(n0 + t);
            __bf16* dst = &s_in[t * 288 + 256];
#pragma unroll
            for (int k = 0; k < 11; ++k) dst[k] = (__bf16)h0[node * 11 + k];
#pragma unroll
            for (int k = 11; k < 32; ++k) dst[k] = (__bf16)0.0f;
        }
        __syncthreads();

        v8f acc = {};
        acc = gemm_kloop<9>(s_in, 288, lane, B1, acc);
#pragma unroll
        for (int v = 0; v < 8; ++v) {
            int M = v + 8 * hh;
            s_m[M * 128 + Ncol] = (__bf16)silu_f(acc[v] + bias1);
        }
        __syncthreads();

        v8f acc2 = {};
        acc2 = gemm_kloop<4>(s_m, 128, lane, B2, acc2);
#pragma unroll
        for (int v = 0; v < 8; ++v) {
            int M = v + 8 * hh;
            size_t idx = (size_t)(n0 + M) * 128 + Ncol;
            float hn = h[idx] + acc2[v] + bias2;   // residual (no outer SiLU)
            h[idx] = hn;
            hbf[idx] = (__bf16)hn;
        }
    }
}

// ---------------------------------------------------------------------------
extern "C" void kernel_launch(void* const* d_in, const int* in_sizes, int n_in,
                              void* d_out, int out_size, void* d_ws, size_t ws_size,
                              hipStream_t stream) {
    const float* h0        = (const float*)d_in[0];
    const float* x         = (const float*)d_in[1];
    const int*   eidx      = (const int*)d_in[2];
    const float* edge_attr = (const float*)d_in[3];
    // d_in[4] node_mask: unused by reference
    const float* edge_mask = (const float*)d_in[5];
    // d_in[6] n_nodes scalar
    const float* emb_w = (const float*)d_in[7];
    const float* emb_b = (const float*)d_in[8];
    const float* ew1   = (const float*)d_in[9];
    const float* eb1   = (const float*)d_in[10];
    const float* ew2   = (const float*)d_in[11];
    const float* eb2   = (const float*)d_in[12];
    const float* nw1   = (const float*)d_in[13];
    const float* nb1   = (const float*)d_in[14];
    const float* nw2   = (const float*)d_in[15];
    const float* nb2   = (const float*)d_in[16];

    const int Nn = 50000, E = 800000, L = 4;
    const int* row = eidx;
    const int* col = eidx + E;

    // workspace carve-up (256B aligned)
    char* wsp = (char*)d_ws;
    auto carve = [&](size_t bytes) -> char* {
        char* p = wsp;
        wsp += (bytes + 255) & ~(size_t)255;
        return p;
    };
    __bf16* hbf    = (__bf16*)carve((size_t)Nn * 128 * 2);
    float*  radial = (float*)carve((size_t)E * 4);
    float*  agg    = (float*)carve((size_t)Nn * 128 * 4);
    const size_t W1F_L = 9 * 8 * 512;   // bf16 elems per layer (K padded to 288)
    const size_t W2F_L = 4 * 8 * 512;   // bf16 elems per layer (K = 128)
    __bf16* ew1f = (__bf16*)carve(L * W1F_L * 2);
    __bf16* ew2f = (__bf16*)carve(L * W2F_L * 2);
    __bf16* nw1f = (__bf16*)carve(L * W1F_L * 2);
    __bf16* nw2f = (__bf16*)carve(L * W2F_L * 2);

    float* h = (float*)d_out;           // running hidden state lives in d_out

    // 1) fragmentize weights into WMMA B-layout (bf16)
    fragw_kernel<<<(int)((L * W1F_L + 255) / 256), 256, 0, stream>>>(ew1, ew1f, 261, 9, L);
    fragw_kernel<<<(int)((L * W2F_L + 255) / 256), 256, 0, stream>>>(ew2, ew2f, 128, 4, L);
    fragw_kernel<<<(int)((L * W1F_L + 255) / 256), 256, 0, stream>>>(nw1, nw1f, 267, 9, L);
    fragw_kernel<<<(int)((L * W2F_L + 255) / 256), 256, 0, stream>>>(nw2, nw2f, 128, 4, L);

    // 2) radial + embedding
    radial_kernel<<<(E + 255) / 256, 256, 0, stream>>>(x, row, col, radial, E);
    embed_kernel<<<(Nn * 128 + 255) / 256, 256, 0, stream>>>(h0, emb_w, emb_b, h, hbf, Nn);

    // 3) layers
    const int edgeWGs = E / 16 / EDGE_TPW;   // 3125
    const int nodeWGs = Nn / 16 / NODE_TPW;  // 625
    for (int l = 0; l < L; ++l) {
        zero_kernel<<<(Nn * 128 + 255) / 256, 256, 0, stream>>>(agg, Nn * 128);
        edge_kernel<<<edgeWGs, 256, 0, stream>>>(hbf, radial, edge_attr, edge_mask,
                                                 row, col,
                                                 ew1f + l * W1F_L, eb1 + l * 128,
                                                 ew2f + l * W2F_L, eb2 + l * 128,
                                                 agg);
        node_kernel<<<nodeWGs, 256, 0, stream>>>(h, agg, h0,
                                                 nw1f + l * W1F_L, nb1 + l * 128,
                                                 nw2f + l * W2F_L, nb2 + l * 128,
                                                 hbf);
    }
}
